// JEGOMamba_75144747811001
// MI455X (gfx1250) — compile-verified, matrix operand order checked by reading, not run on hardware
//
#include <hip/hip_runtime.h>
#include <hip/hip_bf16.h>
#include <math.h>

typedef __bf16 bf16;
typedef __attribute__((ext_vector_type(16))) __bf16 v16bf;
typedef __attribute__((ext_vector_type(8)))  __bf16 v8bf;
typedef __attribute__((ext_vector_type(8)))  float  v8f;

#define AS1 __attribute__((address_space(1)))
#define AS3 __attribute__((address_space(3)))

#define HGRID 48
#define NPIX  2304      // 48*48
#define BATCHN 2
#define HIDC  256
#define DIC   512       // D_INNER
#define DSC   16        // D_STATE
#define LSEQ  4608      // 2*NPIX
#define TOKN  9216      // BATCHN*LSEQ
#define LDXD  544       // D_INNER + 2*D_STATE

// ---- CDNA5 feature detection (device pass only; host pass falls back) -----
#if defined(__has_builtin)
#if __has_builtin(__builtin_amdgcn_global_load_async_to_lds_b128)
#define HAS_ASYNC_LDS 1
#endif
#if __has_builtin(__builtin_amdgcn_tensor_load_to_lds)
#define HAS_TDM 1
#endif
#endif
#ifndef HAS_ASYNC_LDS
#define HAS_ASYNC_LDS 0
#endif
#ifndef HAS_TDM
#define HAS_TDM 0
#endif

#if HAS_TDM
#if defined(__has_include)
#if __has_include(<hip/amd_detail/amd_gfx1250_TDM.h>)
#define TDM_6ARG 1      // therock/clang-23 toolchain: 6-arg builtin
#endif
#endif
#endif
#ifndef TDM_6ARG
#define TDM_6ARG 0
#endif

typedef __attribute__((ext_vector_type(4))) unsigned int u32x4;
typedef __attribute__((ext_vector_type(8))) int i32x8;
typedef __attribute__((ext_vector_type(4))) int i32x4;

__device__ __forceinline__ void wait_asynccnt0()
{
#if defined(__has_builtin) && __has_builtin(__builtin_amdgcn_s_wait_asynccnt)
    __builtin_amdgcn_s_wait_asynccnt(0);
#else
    asm volatile("s_wait_asynccnt 0x0" ::: "memory");
#endif
}

__device__ __forceinline__ void wait_tensorcnt0()
{
#if defined(__has_builtin) && __has_builtin(__builtin_amdgcn_s_wait_tensorcnt)
    __builtin_amdgcn_s_wait_tensorcnt(0);
#else
    asm volatile("s_wait_tensorcnt 0x0" ::: "memory");
#endif
}

// Raw 32-bit LDS byte address of a __shared__ object (addrspacecast -> ptrtoint)
__device__ __forceinline__ unsigned lds_addr_of(const void* p)
{
    return (unsigned)(size_t)(AS3 const void*)p;
}

// ---------------------------------------------------------------------------
// WMMA GEMM: C[M,N] = A[M,K(lda)] * W[N,K]^T (+bias[N]) (+res[M,ldc])
// grid = (ceil(N/128), M/16), block = 128 threads = 4 waves.
// Each wave owns a 16(M)x32(N) slab -> 2 wmma per K-step, A fragment reused.
// A tile (16x32 fp32) staged via GLOBAL_LOAD_ASYNC_TO_LDS (ASYNCcnt).
// W tile (128x32 bf16) staged via the Tensor Data Mover (TENSORcnt); the D#
// tensor dims give hardware zero-fill at the N edge (e.g. N=544).
// ---------------------------------------------------------------------------
__global__ void __launch_bounds__(128)
gemm_wmma_bf16(const float* __restrict__ A, int lda,
               const bf16* __restrict__ W,
               const float* __restrict__ bias,
               const float* __restrict__ res,
               float* __restrict__ C, int ldc,
               int N, int K)
{
    __shared__ __align__(16) float Asf[16 * 32];
#if HAS_TDM
    __shared__ __align__(64) bf16 Ws[128 * 32];
#endif
    const int tid  = threadIdx.x;
    const int wave = tid >> 5;
    const int lane = tid & 31;
    const int half = lane >> 4;
    const int r    = lane & 15;
    const int m0   = blockIdx.y << 4;
    const int nblk = blockIdx.x << 7;
    const int n0   = nblk + (wave << 5);     // this wave's 32-column slab
    const bool act0 = (n0 < N);
    const bool act1 = (n0 + 16 < N);

    v8f acc0 = {0.f, 0.f, 0.f, 0.f, 0.f, 0.f, 0.f, 0.f};
    v8f acc1 = {0.f, 0.f, 0.f, 0.f, 0.f, 0.f, 0.f, 0.f};

    const int rr = tid >> 3;          // staging row 0..15
    const int kk = (tid & 7) << 2;    // staging col 0,4,..,28

    for (int k0 = 0; k0 < K; k0 += 32) {
        const float* gsrc = &A[(size_t)(m0 + rr) * lda + k0 + kk];
        __builtin_prefetch((const void*)(gsrc + 32), 0, 1);   // next K tile
#if HAS_ASYNC_LDS
        __builtin_amdgcn_global_load_async_to_lds_b128(
            (AS1 i32x4*)(size_t)gsrc,
            (AS3 i32x4*)&Asf[tid << 2],
            0, 0);
#else
        *(float4*)&Asf[tid << 2] = *(const float4*)gsrc;
#endif
#if HAS_TDM
        if (wave == 0) {
            // D# group0: count=1 | lds_addr | 57-bit global addr | type=2
            const unsigned long long ga =
                (unsigned long long)(size_t)(W + (size_t)nblk * K + k0);
            u32x4 g0;
            g0.x = 1u;
            g0.y = lds_addr_of(Ws);
            g0.z = (unsigned)ga;
            g0.w = ((unsigned)(ga >> 32) & 0x01FFFFFFu) | (2u << 30);
            // D# group1: data_size=2B; tensor_dim0=K-k0; tensor_dim1=N-nblk;
            // tile 32x128; dim0 stride = K elements.
            const unsigned td0 = (unsigned)(K - k0);
            const unsigned td1 = (unsigned)(N - nblk);
            i32x8 g1 = {0, 0, 0, 0, 0, 0, 0, 0};
            g1[0] = (int)(1u << 16);            // data_size = 2 bytes
            g1[1] = (int)(td0 << 16);           // tensor_dim0 [15:0]
            g1[2] = (int)(td1 << 16);           // tensor_dim1 [15:0]
            g1[3] = (int)(32u << 16);           // tile_dim0 = 32
            g1[4] = 128;                        // tile_dim1 = 128 (tile_dim2=0)
            g1[5] = K;                          // tensor_dim0_stride
            i32x4 z4 = {0, 0, 0, 0};
#if TDM_6ARG
            i32x8 z8 = {0, 0, 0, 0, 0, 0, 0, 0};
            __builtin_amdgcn_tensor_load_to_lds(g0, g1, z4, z4, z8, 0);
#else
            __builtin_amdgcn_tensor_load_to_lds(g0, g1, z4, z4, 0);
#endif
            wait_tensorcnt0();
        }
#endif
#if HAS_ASYNC_LDS
        wait_asynccnt0();
#endif
        __syncthreads();

        // ---- A fragment (documented 16-bit layout), fp32 LDS -> bf16 regs --
        const float* arow = &Asf[r << 5];
        const float4 fa = *(const float4*)(arow + (half << 3));
        const float4 fb = *(const float4*)(arow + (half << 3) + 4);
        const float4 fc = *(const float4*)(arow + 16 + (half << 3));
        const float4 fd = *(const float4*)(arow + 16 + (half << 3) + 4);
        v16bf a;
        a[0]  = (bf16)fa.x; a[1]  = (bf16)fa.y; a[2]  = (bf16)fa.z; a[3]  = (bf16)fa.w;
        a[4]  = (bf16)fb.x; a[5]  = (bf16)fb.y; a[6]  = (bf16)fb.z; a[7]  = (bf16)fb.w;
        a[8]  = (bf16)fc.x; a[9]  = (bf16)fc.y; a[10] = (bf16)fc.z; a[11] = (bf16)fc.w;
        a[12] = (bf16)fd.x; a[13] = (bf16)fd.y; a[14] = (bf16)fd.z; a[15] = (bf16)fd.w;

#if HAS_TDM
        // ---- W fragments from LDS (TDM zero-fills OOB rows) ---------------
        {
            const bf16* w0p = Ws + ((size_t)((wave << 5) + r)) * 32;
            const bf16* w1p = w0p + 16 * 32;
            v8bf b0lo = *(const v8bf*)(w0p + (half << 3));
            v8bf b0hi = *(const v8bf*)(w0p + 16 + (half << 3));
            v16bf b0 = __builtin_shufflevector(b0lo, b0hi,
                0,1,2,3,4,5,6,7,8,9,10,11,12,13,14,15);
            acc0 = __builtin_amdgcn_wmma_f32_16x16x32_bf16(
                false, a, false, b0, (short)0, acc0, false, false);
            v8bf b1lo = *(const v8bf*)(w1p + (half << 3));
            v8bf b1hi = *(const v8bf*)(w1p + 16 + (half << 3));
            v16bf b1 = __builtin_shufflevector(b1lo, b1hi,
                0,1,2,3,4,5,6,7,8,9,10,11,12,13,14,15);
            acc1 = __builtin_amdgcn_wmma_f32_16x16x32_bf16(
                false, a, false, b1, (short)0, acc1, false, false);
        }
#else
        // ---- W fragments straight from global (K-contiguous rows) ---------
        if (act0) {
            const bf16* wrow = W + (size_t)(n0 + r) * K + k0;
            v8bf blo = *(const v8bf*)(wrow + (half << 3));
            v8bf bhi = *(const v8bf*)(wrow + 16 + (half << 3));
            v16bf b0 = __builtin_shufflevector(blo, bhi,
                0,1,2,3,4,5,6,7,8,9,10,11,12,13,14,15);
            acc0 = __builtin_amdgcn_wmma_f32_16x16x32_bf16(
                false, a, false, b0, (short)0, acc0, false, false);
        }
        if (act1) {
            const bf16* wrow = W + (size_t)(n0 + 16 + r) * K + k0;
            v8bf blo = *(const v8bf*)(wrow + (half << 3));
            v8bf bhi = *(const v8bf*)(wrow + 16 + (half << 3));
            v16bf b1 = __builtin_shufflevector(blo, bhi,
                0,1,2,3,4,5,6,7,8,9,10,11,12,13,14,15);
            acc1 = __builtin_amdgcn_wmma_f32_16x16x32_bf16(
                false, a, false, b1, (short)0, acc1, false, false);
        }
#endif
        __syncthreads();
    }

    // ---- epilogue: C/D layout (lane<16: M=j, lane>=16: M=8+j; N=lane%16) --
    if (act0) {
        const int col = n0 + r;
        const float bv = bias ? bias[col] : 0.f;
#pragma unroll
        for (int j = 0; j < 8; ++j) {
            const int row = m0 + (half << 3) + j;
            float v = acc0[j] + bv;
            if (res) v += res[(size_t)row * ldc + col];
            C[(size_t)row * ldc + col] = v;
        }
    }
    if (act1) {
        const int col = n0 + 16 + r;
        const float bv = bias ? bias[col] : 0.f;
#pragma unroll
        for (int j = 0; j < 8; ++j) {
            const int row = m0 + (half << 3) + j;
            float v = acc1[j] + bv;
            if (res) v += res[(size_t)row * ldc + col];
            C[(size_t)row * ldc + col] = v;
        }
    }
}

// ---------------------------------------------------------------------------
// fp32 -> bf16 weight conversion
// ---------------------------------------------------------------------------
__global__ void cvt_bf16_k(const float* __restrict__ in, bf16* __restrict__ out, int n)
{
    const int gid = blockIdx.x * blockDim.x + threadIdx.x;
    if (gid < n) out[gid] = (bf16)in[gid];
}

// pack aggregator conv weights (co,ci,j) -> bf16 (co, j*256+ci)
__global__ void pack_aggw_k(const float* __restrict__ w, bf16* __restrict__ out)
{
    const int gid = blockIdx.x * blockDim.x + threadIdx.x;   // 256*768
    if (gid >= 256 * 768) return;
    const int co = gid / 768;
    const int rk = gid - co * 768;
    const int j  = rk >> 8;
    const int ci = rk & 255;
    out[gid] = (bf16)w[(size_t)co * 768 + ci * 3 + j];
}

// ---------------------------------------------------------------------------
// Directional sequence gather. dir 0=right,1=left,2=down,3=up.
// ---------------------------------------------------------------------------
__global__ void gather_dir_k(const float* __restrict__ f0, const float* __restrict__ f1,
                             float* __restrict__ seq, int dir)
{
    const int gid = blockIdx.x * blockDim.x + threadIdx.x;   // TOKN*64
    const int c4  = gid & 63;
    const int t   = gid >> 6;
    const int b   = t / LSEQ;
    const int pos = t - b * LSEQ;
    int a0 = pos / 96, a1 = pos % 96;
    if (dir == 1 || dir == 3) a1 = 95 - a1;
    int h, w, sel;
    if (dir < 2) { h = a0; sel = (a1 >= 48); w = a1 - sel * 48; }
    else         { w = a0; sel = (a1 >= 48); h = a1 - sel * 48; }
    const float* src = sel ? f1 : f0;
    const float4* s4 = (const float4*)(src + ((size_t)b * NPIX + h * 48 + w) * HIDC);
    ((float4*)(seq + (size_t)t * HIDC))[c4] = s4[c4];
}

// ---------------------------------------------------------------------------
// LayerNorm over HID=256; one wave per token, 8 tokens per block.
// ---------------------------------------------------------------------------
__global__ void __launch_bounds__(256)
layernorm_k(const float* __restrict__ x, const float* __restrict__ g,
            const float* __restrict__ bta, float* __restrict__ y)
{
    const int wave = threadIdx.x >> 5;
    const int lane = threadIdx.x & 31;
    const int t = blockIdx.x * 8 + wave;
    const float* xr = x + (size_t)t * HIDC;
    float v[8];
    float s = 0.f;
#pragma unroll
    for (int j = 0; j < 8; ++j) { v[j] = xr[lane + (j << 5)]; s += v[j]; }
#pragma unroll
    for (int o = 16; o; o >>= 1) s += __shfl_xor(s, o, 32);
    const float mu = s * (1.f / HIDC);
    float q = 0.f;
#pragma unroll
    for (int j = 0; j < 8; ++j) { const float dd = v[j] - mu; q += dd * dd; }
#pragma unroll
    for (int o = 16; o; o >>= 1) q += __shfl_xor(q, o, 32);
    const float inv = rsqrtf(q * (1.f / HIDC) + 1e-5f);
    float* yr = y + (size_t)t * HIDC;
#pragma unroll
    for (int j = 0; j < 8; ++j) {
        const int c = lane + (j << 5);
        yr[c] = (v[j] - mu) * inv * g[c] + bta[c];
    }
}

// ---------------------------------------------------------------------------
// Depthwise causal conv (k=4) on xi (= low half of xz, ld 1024) + SiLU -> xs
// ---------------------------------------------------------------------------
__global__ void dwconv_silu_k(const float* __restrict__ xz,
                              const float* __restrict__ cw,
                              const float* __restrict__ cb,
                              float* __restrict__ xs)
{
    const int gid = blockIdx.x * blockDim.x + threadIdx.x;   // TOKN*DIC
    const int d = gid & (DIC - 1);
    const int t = gid >> 9;
    const int b = t / LSEQ;
    const int l = t - b * LSEQ;
    float acc = cb[d];
#pragma unroll
    for (int j = 0; j < 4; ++j) {
        const int ls = l + j - 3;
        if (ls >= 0)
            acc += cw[(d << 2) + j] * xz[((size_t)(b * LSEQ + ls) << 10) + d];
    }
    acc = acc / (1.f + __expf(-acc));    // SiLU
    xs[(size_t)gid] = acc;
}

__global__ void softplus_k(float* __restrict__ x, int n)
{
    const int gid = blockIdx.x * blockDim.x + threadIdx.x;
    if (gid < n) {
        const float v = x[gid];
        x[gid] = (v > 20.f) ? v : log1pf(__expf(v));
    }
}

__global__ void gate_silu_k(float* __restrict__ y, const float* __restrict__ xz)
{
    const int gid = blockIdx.x * blockDim.x + threadIdx.x;   // TOKN*DIC
    const int d = gid & (DIC - 1);
    const int t = gid >> 9;
    const float z = xz[((size_t)t << 10) + DIC + d];
    y[gid] *= z / (1.f + __expf(-z));
}

__global__ void gelu_mul_k(float* __restrict__ a, const float* __restrict__ b, int n)
{
    const int gid = blockIdx.x * blockDim.x + threadIdx.x;
    if (gid < n) {
        const float v = a[gid];
        a[gid] = 0.5f * v * (1.f + erff(v * 0.70710678118654752f)) * b[gid];
    }
}

// ---------------------------------------------------------------------------
// Selective scan, parallel over (batch, channel, state); 16-lane groups do
// the per-step y reduction via shfl_xor. B/C rows chunk-staged in LDS.
// grid = B*DI*DS/256 = 64.
// ---------------------------------------------------------------------------
__global__ void __launch_bounds__(256)
scan_k(const float* __restrict__ xs, const float* __restrict__ dlt,
       const float* __restrict__ xdbl, const float* __restrict__ A_log,
       const float* __restrict__ Dp, float* __restrict__ y)
{
    __shared__ float bc[64][32];
    const int gid = blockIdx.x * 256 + threadIdx.x;
    const int s = gid & 15;
    const int d = (gid >> 4) & (DIC - 1);
    const int b = gid >> 13;                 // DI*DS = 8192 per batch
    const float Aval = -__expf(A_log[(d << 4) + s]);
    const float Dv = Dp[d];
    const float* base = xdbl + (size_t)b * LSEQ * LDXD + DIC;  // cols 512..543
    float h = 0.f;
    for (int t0 = 0; t0 < LSEQ; t0 += 64) {
        __syncthreads();
        for (int i = threadIdx.x; i < 64 * 32; i += 256)
            bc[i >> 5][i & 31] = base[(size_t)(t0 + (i >> 5)) * LDXD + (i & 31)];
        __syncthreads();
        for (int tt = 0; tt < 64; ++tt) {
            const int t = t0 + tt;
            const size_t off = ((size_t)(b * LSEQ + t) << 9) + d;
            const float dlv = dlt[off];
            const float xv  = xs[off];
            h = __expf(dlv * Aval) * h + (dlv * xv) * bc[tt][s];
            float c = h * bc[tt][16 + s];
#pragma unroll
            for (int o = 8; o; o >>= 1) c += __shfl_xor(c, o, 16);
            if (s == 0) y[off] = c + Dv * xv;
        }
    }
}

// ---------------------------------------------------------------------------
// Merge the four processed directional sequences (inverse reshapes + average)
// ---------------------------------------------------------------------------
__global__ void merge_dirs_k(const float* __restrict__ so, float* __restrict__ merged)
{
    const int gid = blockIdx.x * blockDim.x + threadIdx.x;   // TOKN*64
    const int c4 = gid & 63;
    const int t  = gid >> 6;
    const int b  = t / LSEQ;
    const int l  = t - b * LSEQ;
    const int n  = (l < NPIX) ? l : (l - NPIX);
    const bool second = (l >= NPIX);
    const int h = n / 48, w = n - (n / 48) * 48;
    int ir, il, idn, iu;
    if (!second) {
        ir = h * 96 + w;       il = h * 96 + (95 - w);
        idn = w * 96 + h;      iu = w * 96 + (95 - h);
    } else {
        ir = h * 96 + 48 + w;  il = h * 96 + (47 - w);
        idn = w * 96 + 48 + h; iu = w * 96 + (47 - h);
    }
    const size_t stride = (size_t)TOKN * HIDC;
    const size_t brow = (size_t)b * LSEQ;
    const float4 vr = ((const float4*)(so + 0 * stride + (brow + ir)  * HIDC))[c4];
    const float4 vl = ((const float4*)(so + 1 * stride + (brow + il)  * HIDC))[c4];
    const float4 vd = ((const float4*)(so + 2 * stride + (brow + idn) * HIDC))[c4];
    const float4 vu = ((const float4*)(so + 3 * stride + (brow + iu)  * HIDC))[c4];
    float4 o;
    o.x = 0.25f * (vr.x + vl.x + vd.x + vu.x);
    o.y = 0.25f * (vr.y + vl.y + vd.y + vu.y);
    o.z = 0.25f * (vr.z + vl.z + vd.z + vu.z);
    o.w = 0.25f * (vr.w + vl.w + vd.w + vu.w);
    ((float4*)(merged + (size_t)t * HIDC))[c4] = o;
}

// im2col-style pack for k=3 "same" conv: acat[t, j*256+ci] = x[l+j-1, ci]
__global__ void pack_acat_k(const float* __restrict__ x, float* __restrict__ acat)
{
    const int gid = blockIdx.x * blockDim.x + threadIdx.x;   // TOKN*192 float4
    const int c4 = gid % 192;
    const int t  = gid / 192;
    const int b  = t / LSEQ;
    const int l  = t - b * LSEQ;
    const int j  = c4 >> 6;
    const int cc = c4 & 63;
    const int ls = l + j - 1;
    float4 v = {0.f, 0.f, 0.f, 0.f};
    if (ls >= 0 && ls < LSEQ)
        v = ((const float4*)(x + ((size_t)(b * LSEQ + ls)) * HIDC))[cc];
    ((float4*)(acat + (size_t)t * 768))[c4] = v;
}

// (B,2N,256) -> e0 (B,N,256) ++ e1 (B,N,256)
__global__ void reorder_out_k(const float* __restrict__ proj, float* __restrict__ out)
{
    const int gid = blockIdx.x * blockDim.x + threadIdx.x;   // TOKN*64
    const int c4 = gid & 63;
    const int t  = gid >> 6;
    const int b  = t / LSEQ;
    const int l  = t - b * LSEQ;
    const size_t drow = (l < NPIX)
        ? ((size_t)b * NPIX + l)
        : ((size_t)BATCHN * NPIX + (size_t)b * NPIX + (l - NPIX));
    ((float4*)(out + drow * HIDC))[c4] = ((const float4*)(proj + (size_t)t * HIDC))[c4];
}

// ---------------------------------------------------------------------------
// Host orchestration
// ---------------------------------------------------------------------------
extern "C" void kernel_launch(void* const* d_in, const int* in_sizes, int n_in,
                              void* d_out, int out_size, void* d_ws, size_t ws_size,
                              hipStream_t stream)
{
    (void)in_sizes; (void)n_in; (void)out_size; (void)ws_size;

    // Pytree leaf order (dicts by sorted key, list in order):
    const float* F0   = (const float*)d_in[0];
    const float* F1   = (const float*)d_in[1];
    const float* AGB[3] = {(const float*)d_in[2], (const float*)d_in[3], (const float*)d_in[4]};
    const float* AGW[3] = {(const float*)d_in[5], (const float*)d_in[6], (const float*)d_in[7]};
    auto blk = [&](int i, int j) { return (const float*)d_in[8 + 11 * i + j]; };
    const float* IN_B  = (const float*)d_in[52];
    const float* IN_W  = (const float*)d_in[53];
    const float* OUT_B = (const float*)d_in[54];
    const float* OUT_W = (const float*)d_in[55];

    char* p = (char*)d_ws;
    auto alloc = [&](size_t bytes) -> char* {
        char* r = p; p += (bytes + 255) & ~(size_t)255; return r;
    };
    float* f0f1    = (float*)alloc((size_t)TOKN * HIDC * 4);       // f0 ++ f1
    float* seq_in  = (float*)alloc((size_t)TOKN * HIDC * 4);
    float* seq_out = (float*)alloc((size_t)4 * TOKN * HIDC * 4);
    float* xn      = (float*)alloc((size_t)TOKN * HIDC * 4);
    float* xz      = (float*)alloc((size_t)TOKN * 1024 * 4);
    float* xs      = (float*)alloc((size_t)TOKN * DIC * 4);
    float* xdbl    = (float*)alloc((size_t)TOKN * LDXD * 4);
    float* dl      = (float*)alloc((size_t)TOKN * DIC * 4);
    float* yb      = (float*)alloc((size_t)TOKN * DIC * 4);
    bf16* wbf_in   = (bf16*)alloc((size_t)HIDC * HIDC * 2);
    bf16* wbf_out  = (bf16*)alloc((size_t)HIDC * HIDC * 2);
    bf16* wbf_ag0  = (bf16*)alloc((size_t)HIDC * 768 * 2);
    bf16* wbf_ag1  = (bf16*)alloc((size_t)HIDC * 768 * 2);
    bf16* wbf_ag2  = (bf16*)alloc((size_t)HIDC * 768 * 2);
    bf16* wbf_win  = (bf16*)alloc((size_t)1024 * HIDC * 2);
    bf16* wbf_wx   = (bf16*)alloc((size_t)LDXD * DIC * 2);
    bf16* wbf_wdt  = (bf16*)alloc((size_t)DIC * DIC * 2);
    bf16* wbf_wout = (bf16*)alloc((size_t)HIDC * DIC * 2);
    // Aliases for post-block stages (those buffers are free by then):
    float* merged = xn;
    float* acat   = xz;
    float* t1     = xs;
    float* t2     = dl;
    float* proj   = yb;
    bf16* wbf_ag[3] = {wbf_ag0, wbf_ag1, wbf_ag2};

    auto gemm = [&](const float* A, int lda, const bf16* W, const float* bias,
                    const float* res, float* C, int ldc, int M, int N, int K) {
        dim3 g((N + 127) / 128, M / 16, 1);
        gemm_wmma_bf16<<<g, dim3(128), 0, stream>>>(A, lda, W, bias, res, C, ldc, N, K);
    };
    auto cvt = [&](const float* in, bf16* out, int n) {
        cvt_bf16_k<<<(n + 255) / 256, 256, 0, stream>>>(in, out, n);
    };

    // ---- input projections --------------------------------------------------
    cvt(IN_W, wbf_in, HIDC * HIDC);
    gemm(F0, HIDC, wbf_in, IN_B, nullptr, f0f1, HIDC, BATCHN * NPIX, HIDC, HIDC);
    gemm(F1, HIDC, wbf_in, IN_B, nullptr, f0f1 + (size_t)BATCHN * NPIX * HIDC,
         HIDC, BATCHN * NPIX, HIDC, HIDC);

    // ---- four directional Mamba blocks -------------------------------------
    for (int dir = 0; dir < 4; ++dir) {
        gather_dir_k<<<TOKN * 64 / 256, 256, 0, stream>>>(
            f0f1, f0f1 + (size_t)BATCHN * NPIX * HIDC, seq_in, dir);
        cvt(blk(dir, 8),  wbf_win,  1024 * HIDC);
        cvt(blk(dir, 10), wbf_wx,   LDXD * DIC);
        cvt(blk(dir, 7),  wbf_wdt,  DIC * DIC);
        cvt(blk(dir, 9),  wbf_wout, HIDC * DIC);

        layernorm_k<<<TOKN / 8, 256, 0, stream>>>(seq_in, blk(dir, 6), blk(dir, 5), xn);
        gemm(xn, HIDC, wbf_win, nullptr, nullptr, xz, 1024, TOKN, 1024, HIDC);
        dwconv_silu_k<<<TOKN * DIC / 256, 256, 0, stream>>>(xz, blk(dir, 4), blk(dir, 3), xs);
        gemm(xs, DIC, wbf_wx, nullptr, nullptr, xdbl, LDXD, TOKN, LDXD, DIC);
        gemm(xdbl, LDXD, wbf_wdt, blk(dir, 2), nullptr, dl, DIC, TOKN, DIC, DIC);
        softplus_k<<<TOKN * DIC / 256, 256, 0, stream>>>(dl, TOKN * DIC);
        scan_k<<<BATCHN * DIC * DSC / 256, 256, 0, stream>>>(
            xs, dl, xdbl, blk(dir, 0), blk(dir, 1), yb);
        gate_silu_k<<<TOKN * DIC / 256, 256, 0, stream>>>(yb, xz);
        gemm(yb, DIC, wbf_wout, nullptr, seq_in,
             seq_out + (size_t)dir * TOKN * HIDC, HIDC, TOKN, HIDC, DIC);
    }

    // ---- merge + aggregator -------------------------------------------------
    merge_dirs_k<<<TOKN * 64 / 256, 256, 0, stream>>>(seq_out, merged);
    for (int i = 0; i < 3; ++i)
        pack_aggw_k<<<(256 * 768 + 255) / 256, 256, 0, stream>>>(AGW[i], wbf_ag[i]);
    pack_acat_k<<<TOKN * 192 / 256, 256, 0, stream>>>(merged, acat);
    gemm(acat, 768, wbf_ag[0], AGB[0], nullptr, t1, HIDC, TOKN, HIDC, 768);
    gemm(acat, 768, wbf_ag[1], AGB[1], nullptr, t2, HIDC, TOKN, HIDC, 768);
    gelu_mul_k<<<TOKN * HIDC / 256, 256, 0, stream>>>(t1, t2, TOKN * HIDC);
    pack_acat_k<<<TOKN * 192 / 256, 256, 0, stream>>>(t1, acat);
    gemm(acat, 768, wbf_ag[2], AGB[2], merged, t2, HIDC, TOKN, HIDC, 768);

    // ---- output projection --------------------------------------------------
    cvt(OUT_W, wbf_out, HIDC * HIDC);
    gemm(t2, HIDC, wbf_out, OUT_B, nullptr, proj, HIDC, TOKN, HIDC, HIDC);
    reorder_out_k<<<TOKN * 64 / 256, 256, 0, stream>>>(proj, (float*)d_out);
}
